// GCN_10943576670375
// MI455X (gfx1250) — compile-verified, matrix operand order checked by reading, not run on hardware
//
#include <hip/hip_runtime.h>
#include <hip/hip_bf16.h>

typedef __attribute__((ext_vector_type(16))) _Float16 v16h;
typedef __attribute__((ext_vector_type(8)))  float    v8f;

// ---------------------------------------------------------------- degree prep
__global__ void gcn_deg_init(float* __restrict__ deg, int n) {
    int i = blockIdx.x * blockDim.x + threadIdx.x;
    if (i < n) deg[i] = 1.0f;  // self-loop contributes 1
}

__global__ void gcn_deg_accum(float* __restrict__ deg, const int* __restrict__ dst, int E) {
    int i = blockIdx.x * blockDim.x + threadIdx.x;
    if (i < E) atomicAdd(&deg[dst[i]], 1.0f);
}

__global__ void gcn_deg_rsqrt(float* __restrict__ deg, int n) {
    int i = blockIdx.x * blockDim.x + threadIdx.x;
    if (i < n) deg[i] = rsqrtf(deg[i]);  // deg >= 1 always (self-loops)
}

// ---------------------------------------------------------------- B packing
// Convert W[K,N] (f32, row-major) into f16 WMMA B-fragments laid out so each
// (k0,tile_n) fragment is a contiguous 1KB block: Bp[frag][lane][e], where
// frag = (k0/32)*(N/16) + tn, col = tn*16 + (lane&15),
// K = k0 + (lane>>4)*8 + (e&7) + (e>=8 ? 16 : 0).
__global__ void gcn_pack_b(const float* __restrict__ B, _Float16* __restrict__ Bp,
                           int K, int N) {
    int i = blockIdx.x * blockDim.x + threadIdx.x;      // over K*N packed halves
    if (i >= K * N) return;
    int e    = i & 15;
    int lane = (i >> 4) & 31;
    int frag = i >> 9;                                   // 512 halves per fragment
    int nt   = N >> 4;
    int k0   = (frag / nt) << 5;
    int tn   = frag % nt;
    int col  = tn * 16 + (lane & 15);
    int kk   = k0 + (lane >> 4) * 8 + (e & 7) + ((e >> 3) << 4);
    Bp[i] = (_Float16)B[kk * N + col];
}

// ---------------------------------------------------------------- WMMA GEMM
// C[M,N] = op(A[M,K]) @ B[K,N]; A f32 (converted inline), B pre-packed f16,
// f32 accumulate. One wave computes a 16x64 strip: one A fragment shared by
// 4 WMMAs per 32-deep K step; each B fragment is 2x global_load_b128.
template <bool RELU>
__global__ void gcn_gemm_wmma(const float* __restrict__ A, const _Float16* __restrict__ Bp,
                              float* __restrict__ C, int M, int K, int N) {
    const int lane = threadIdx.x & 31;
    const int wid  = threadIdx.x >> 5;
    const int ntg  = N >> 6;                       // 64-col groups
    const int nt   = N >> 4;                       // 16-col tiles
    const int wg   = blockIdx.x * (blockDim.x >> 5) + wid;
    const int tm   = wg / ntg;                     // 16-row tile index
    const int tg   = wg % ntg;                     // 64-col group index
    if (tm * 16 >= M) return;

    const int g   = lane >> 4;                     // half-wave select
    const int l16 = lane & 15;
    int arow = tm * 16 + l16;
    if (arow >= M) arow = M - 1;                   // clamp (tail stores guarded)

    v8f acc0 = {}, acc1 = {}, acc2 = {}, acc3 = {};
    for (int k0 = 0; k0 < K; k0 += 32) {
        // A fragment: element e -> K = k0 + g*8 + (e&7) + (e>=8 ? 16 : 0)
        const float4* ap4 = (const float4*)(A + arow * K + k0 + g * 8);
        float4 a0 = ap4[0], a1 = ap4[1], a2 = ap4[4], a3 = ap4[5];
        if (RELU) {
            a0.x = fmaxf(a0.x, 0.f); a0.y = fmaxf(a0.y, 0.f); a0.z = fmaxf(a0.z, 0.f); a0.w = fmaxf(a0.w, 0.f);
            a1.x = fmaxf(a1.x, 0.f); a1.y = fmaxf(a1.y, 0.f); a1.z = fmaxf(a1.z, 0.f); a1.w = fmaxf(a1.w, 0.f);
            a2.x = fmaxf(a2.x, 0.f); a2.y = fmaxf(a2.y, 0.f); a2.z = fmaxf(a2.z, 0.f); a2.w = fmaxf(a2.w, 0.f);
            a3.x = fmaxf(a3.x, 0.f); a3.y = fmaxf(a3.y, 0.f); a3.z = fmaxf(a3.z, 0.f); a3.w = fmaxf(a3.w, 0.f);
        }
        v16h a;
        a[0]  = (_Float16)a0.x; a[1]  = (_Float16)a0.y; a[2]  = (_Float16)a0.z; a[3]  = (_Float16)a0.w;
        a[4]  = (_Float16)a1.x; a[5]  = (_Float16)a1.y; a[6]  = (_Float16)a1.z; a[7]  = (_Float16)a1.w;
        a[8]  = (_Float16)a2.x; a[9]  = (_Float16)a2.y; a[10] = (_Float16)a2.z; a[11] = (_Float16)a2.w;
        a[12] = (_Float16)a3.x; a[13] = (_Float16)a3.y; a[14] = (_Float16)a3.z; a[15] = (_Float16)a3.w;

        // 4 pre-packed B fragments, each a contiguous 1KB block (2x b128/lane)
        const _Float16* bbase = Bp + ((k0 >> 5) * nt + tg * 4) * 512 + lane * 16;
        v16h b0 = *(const v16h*)(bbase);
        v16h b1 = *(const v16h*)(bbase + 512);
        v16h b2 = *(const v16h*)(bbase + 1024);
        v16h b3 = *(const v16h*)(bbase + 1536);

        acc0 = __builtin_amdgcn_wmma_f32_16x16x32_f16(false, a, false, b0, (short)0, acc0, false, false);
        acc1 = __builtin_amdgcn_wmma_f32_16x16x32_f16(false, a, false, b1, (short)0, acc1, false, false);
        acc2 = __builtin_amdgcn_wmma_f32_16x16x32_f16(false, a, false, b2, (short)0, acc2, false, false);
        acc3 = __builtin_amdgcn_wmma_f32_16x16x32_f16(false, a, false, b3, (short)0, acc3, false, false);
    }
    // D layout: VGPR r -> row = tm*16 + r + 8*g, col = tg*64 + l16 + t*16
    const int cbase = (tm * 16 + 8 * g) * N + tg * 64 + l16;
    if (tm * 16 + 16 <= M) {                       // fast path: no guards
#pragma unroll
        for (int r = 0; r < 8; ++r) {
            float* cp = C + cbase + r * N;
            cp[0]  = acc0[r];
            cp[16] = acc1[r];
            cp[32] = acc2[r];
            cp[48] = acc3[r];
        }
    } else {                                       // tail tile (unused when M%16==0)
#pragma unroll
        for (int r = 0; r < 8; ++r) {
            int m = tm * 16 + r + 8 * g;
            if (m < M) {
                float* cp = C + cbase + r * N;
                cp[0]  = acc0[r];
                cp[16] = acc1[r];
                cp[32] = acc2[r];
                cp[48] = acc3[r];
            }
        }
    }
}

// ------------------------------------------------ self-loop + bias init (out = xw*dinv^2 + b)
__global__ void gcn_self_bias(const float* __restrict__ xw, const float* __restrict__ dinv,
                              const float* __restrict__ bias, float* __restrict__ out,
                              int nrows, int fshift) {
    int i = blockIdx.x * blockDim.x + threadIdx.x;
    if (i >= (nrows << fshift)) return;
    int r = i >> fshift;
    int f = i & ((1 << fshift) - 1);
    float di = dinv[r];
    out[i] = xw[i] * di * di + bias[f];
}

// ------------------------------------------------ per-edge scatter: one wave per edge
template <int F>
__global__ void gcn_edge_scatter(const float* __restrict__ sf, float* __restrict__ df,
                                 const int* __restrict__ src, const int* __restrict__ dst,
                                 const float* __restrict__ dinv, int E) {
    const int wid  = threadIdx.x >> 5;
    const int lane = threadIdx.x & 31;
    const int e = blockIdx.x * (blockDim.x >> 5) + wid;
    if (e >= E) return;
    const int s = src[e], d = dst[e];
    const float nrm = dinv[s] * dinv[d];
    if (F == 128) {
        float4 v = ((const float4*)(sf + s * 128))[lane];   // 512B coalesced gather
        float* dp = df + d * 128 + lane * 4;
        atomicAdd(dp + 0, v.x * nrm);
        atomicAdd(dp + 1, v.y * nrm);
        atomicAdd(dp + 2, v.z * nrm);
        atomicAdd(dp + 3, v.w * nrm);
    } else {                                                // F == 64
        float2 v = ((const float2*)(sf + s * 64))[lane];
        float* dp = df + d * 64 + lane * 2;
        atomicAdd(dp + 0, v.x * nrm);
        atomicAdd(dp + 1, v.y * nrm);
    }
}

// ------------------------------------------------ log_softmax over 64 classes, wave per row
__global__ void gcn_log_softmax64(float* __restrict__ out, int nrows) {
    const int wid  = threadIdx.x >> 5;
    const int lane = threadIdx.x & 31;
    const int row = blockIdx.x * (blockDim.x >> 5) + wid;
    if (row >= nrows) return;
    float* p = out + row * 64;
    float a = p[lane], b = p[lane + 32];
    float m = fmaxf(a, b);
#pragma unroll
    for (int o = 16; o > 0; o >>= 1) m = fmaxf(m, __shfl_xor(m, o, 32));
    float s = expf(a - m) + expf(b - m);
#pragma unroll
    for (int o = 16; o > 0; o >>= 1) s += __shfl_xor(s, o, 32);
    float lse = m + logf(s);
    p[lane]      = a - lse;
    p[lane + 32] = b - lse;
}

// ---------------------------------------------------------------- launcher
extern "C" void kernel_launch(void* const* d_in, const int* in_sizes, int n_in,
                              void* d_out, int out_size, void* d_ws, size_t ws_size,
                              hipStream_t stream) {
    const float* x  = (const float*)d_in[0];
    const float* W1 = (const float*)d_in[1];
    const float* b1 = (const float*)d_in[2];
    const float* W2 = (const float*)d_in[3];
    const float* b2 = (const float*)d_in[4];
    const int*   ei = (const int*)d_in[5];

    const int hid  = in_sizes[2];              // 128
    const int ncls = in_sizes[4];              // 64
    const int inF  = in_sizes[1] / hid;        // 128
    const int N    = in_sizes[0] / inF;        // 100000
    const int E    = in_sizes[5] / 2;          // 1600000
    const int* src = ei;
    const int* dst = ei + E;

    // workspace carve-out (512B aligned)
    char* ws = (char*)d_ws;
    size_t off = 0;
    auto carve = [&](size_t bytes) -> char* {
        off = (off + 511) & ~(size_t)511;
        char* p = ws + off;
        off += bytes;
        return p;
    };
    float*    dinv = (float*)carve((size_t)N * sizeof(float));
    float*    XW1  = (float*)carve((size_t)N * hid * sizeof(float));
    float*    H1   = (float*)carve((size_t)N * hid * sizeof(float));
    float*    XW2  = (float*)carve((size_t)N * ncls * sizeof(float));
    _Float16* W1p  = (_Float16*)carve((size_t)inF * hid * sizeof(_Float16));
    _Float16* W2p  = (_Float16*)carve((size_t)hid * ncls * sizeof(_Float16));
    float*    out  = (float*)d_out;            // [N, ncls]

    const int TB = 256;
    // 0) pack weights into f16 WMMA B-fragment layout (tiny, stays in cache)
    gcn_pack_b<<<(inF * hid + TB - 1) / TB, TB, 0, stream>>>(W1, W1p, inF, hid);
    gcn_pack_b<<<(hid * ncls + TB - 1) / TB, TB, 0, stream>>>(W2, W2p, hid, ncls);

    // 1) degrees -> dinv
    gcn_deg_init<<<(N + TB - 1) / TB, TB, 0, stream>>>(dinv, N);
    gcn_deg_accum<<<(E + TB - 1) / TB, TB, 0, stream>>>(dinv, dst, E);
    gcn_deg_rsqrt<<<(N + TB - 1) / TB, TB, 0, stream>>>(dinv, N);

    // 2) XW1 = X @ W1   (WMMA; 4 waves/block; each wave owns a 16x64 strip)
    {
        int waves = ((N + 15) / 16) * (hid / 64);
        gcn_gemm_wmma<false><<<(waves + 3) / 4, 128, 0, stream>>>(x, W1p, XW1, N, inF, hid);
    }
    // 3) H1 = self-loop + bias, then edge scatter-add
    {
        int total = N * hid;
        gcn_self_bias<<<(total + TB - 1) / TB, TB, 0, stream>>>(XW1, dinv, b1, H1, N, 7);
        gcn_edge_scatter<128><<<(E + 7) / 8, TB, 0, stream>>>(XW1, H1, src, dst, dinv, E);
    }
    // 4) XW2 = relu(H1) @ W2   (ReLU fused into A-fragment load)
    {
        int waves = ((N + 15) / 16) * (ncls / 64);
        gcn_gemm_wmma<true><<<(waves + 3) / 4, 128, 0, stream>>>(H1, W2p, XW2, N, hid, ncls);
    }
    // 5) out = self-loop + bias, then edge scatter-add (into d_out)
    {
        int total = N * ncls;
        gcn_self_bias<<<(total + TB - 1) / TB, TB, 0, stream>>>(XW2, dinv, b2, out, N, 6);
        gcn_edge_scatter<64><<<(E + 7) / 8, TB, 0, stream>>>(XW2, out, src, dst, dinv, E);
    }
    // 6) log_softmax rows (64 classes), in place on d_out
    gcn_log_softmax64<<<(N + 7) / 8, TB, 0, stream>>>(out, N);
}
